// RPN_66666482368741
// MI455X (gfx1250) — compile-verified, matrix operand order checked by reading, not run on hardware
//
#include <hip/hip_runtime.h>
#include <hip/hip_bf16.h>

// ---------------------------------------------------------------------------
// Problem constants (match reference)
// ---------------------------------------------------------------------------
#define BB 8
#define KA 9
#define NGT 64
#define POSN 32768     // B*H*W
#define KHW 36864      // KA*H*W
#define POS_T 0.7f
#define NEG_T 0.3f

// d_out flat layout (floats, concatenated in return order)
#define CLS_OFF   0
#define MOVED_OFF 294912
#define POSL_OFF  1474560
#define NEGL_OFF  1769472
#define HGT_OFF   2064384

// workspace byte offsets (all 64B aligned)
#define WS_BW     ((size_t)0)                 // conv weights bf16, tile-packed
#define WS_WC     ((size_t)4718592)           // head weights bf16 [48][512] (transposed)
#define WS_INTER  ((size_t)4767744)           // inter bf16 [32768][512]
#define WS_BEST   ((size_t)38322176)          // int[512] winner anchor per (b,gt)
#define WS_REGT   ((size_t)38326272)          // float [32768][36] raw reg

typedef __attribute__((ext_vector_type(16))) __bf16 v16bf;
typedef __attribute__((ext_vector_type(8)))  __bf16 v8bf;
typedef __attribute__((ext_vector_type(2)))  __bf16 v2bf;
typedef __attribute__((ext_vector_type(8)))  float  v8f;

// async copy of one 16B chunk: global -> LDS (CDNA5 GLOBAL_LOAD_ASYNC_TO_LDS_B128)
static __device__ __forceinline__ void async_g2l_b128(const void* gptr, void* lptr) {
    unsigned lds_off = (unsigned)(uintptr_t)lptr;            // low 32 bits = LDS offset
    unsigned long long ga = (unsigned long long)(uintptr_t)gptr;
    asm volatile("global_load_async_to_lds_b128 %0, %1, off"
                 :: "v"(lds_off), "v"(ga) : "memory");
}
static __device__ __forceinline__ void wait_async0() {
    asm volatile("s_wait_asynccnt 0x0" ::: "memory");
}

static __device__ __forceinline__ v16bf load_frag(const __bf16* p) {
    v8bf lo = *(const v8bf*)p;
    v8bf hi = *(const v8bf*)(p + 16);
    return __builtin_shufflevector(lo, hi, 0,1,2,3,4,5,6,7,8,9,10,11,12,13,14,15);
}

static __device__ __forceinline__ float iou_xywh(float ax, float ay, float aw, float ah,
                                                 float bx, float by, float bw, float bh) {
    float ax1 = ax - aw * 0.5f, ay1 = ay - ah * 0.5f;
    float ax2 = ax + aw * 0.5f, ay2 = ay + ah * 0.5f;
    float bx1 = bx - bw * 0.5f, by1 = by - bh * 0.5f;
    float bx2 = bx + bw * 0.5f, by2 = by + bh * 0.5f;
    float iw = fmaxf(fminf(ax2, bx2) - fmaxf(ax1, bx1), 0.0f);
    float ih = fmaxf(fminf(ay2, by2) - fmaxf(ay1, by1), 0.0f);
    float inter = iw * ih;
    float uni = aw * ah + bw * bh - inter;
    return inter / fmaxf(uni, 1e-6f);
}

// ---------------------------------------------------------------------------
// K0: repack weights to bf16.
//  Bpk: conv weights packed per (kk, ic-block) tile in the exact LDS [n][32]
//       image: Bpk[((kk*16+icb)*512 + oc)*32 + t], ic = icb*32 + t
//  WcT: head weights transposed [n=48][k=512], channels >=45 zero padded
// ---------------------------------------------------------------------------
__global__ __launch_bounds__(256) void rpn_prep_weights(
    const float* __restrict__ w_inter, const float* __restrict__ w_cls,
    const float* __restrict__ w_reg, __bf16* __restrict__ Bpk, __bf16* __restrict__ WcT) {
    int i = blockIdx.x * 256 + threadIdx.x;
    const int NBW = 4608 * 512;
    if (i < NBW) {
        int t   = i & 31;
        int oc  = (i >> 5) & 511;
        int blk = i >> 14;           // kk*16 + icb
        int icb = blk & 15;
        int kk  = blk >> 4;
        int ic  = icb * 32 + t;
        Bpk[i] = (__bf16)w_inter[(oc * 512 + ic) * 9 + kk];
    } else {
        int j = i - NBW;
        if (j < 48 * 512) {
            int n = j >> 9, k = j & 511;
            float v = 0.0f;
            if (n < 9)       v = w_cls[n * 512 + k];
            else if (n < 45) v = w_reg[(n - 9) * 512 + k];
            WcT[j] = (__bf16)v;
        }
    }
}

// ---------------------------------------------------------------------------
// K1: 3x3 conv 512->512 + ReLU, implicit GEMM, bf16 WMMA, double-buffered LDS,
//     async weight staging, branch-free halo gather.
//   grid (256, 4): x = position tile (128 positions), y = oc tile (128 oc)
// ---------------------------------------------------------------------------
__global__ __launch_bounds__(256) void rpn_conv3_wmma(
    const float* __restrict__ feat, const __bf16* __restrict__ Bpk,
    const float* __restrict__ b_inter, __bf16* __restrict__ inter_t) {
    __shared__ __attribute__((aligned(16))) __bf16 As[2][128 * 40];  // [m][32] +8 pad
    __shared__ __attribute__((aligned(16))) __bf16 Bs[2][128 * 40];  // [n][32] +8 pad

    const int tid  = threadIdx.x;
    const int lane = tid & 31;
    const int wid  = tid >> 5;
    const int wm   = wid >> 2;      // 0..1 -> 64 positions per wave
    const int wn   = wid & 3;       // 0..3 -> 32 oc per wave
    const int m0   = blockIdx.x * 128;
    const int oc0  = blockIdx.y * 128;
    const int koff = (lane & 16) ? 8 : 0;
    const int l15  = lane & 15;

    v8f acc[4][2];
    #pragma unroll
    for (int a = 0; a < 4; ++a)
        #pragma unroll
        for (int b = 0; b < 2; ++b)
            #pragma unroll
            for (int r = 0; r < 8; ++r) acc[a][b][r] = 0.0f;

    // ---- A stage: 128 pos x 32 ic tile, branch-free (clamp + select), packed
    //      pairs along k so each pair is one v_cvt_pk_bf16_f32 + ds_store_b32.
    auto stageA = [&](int s, __bf16* Ab) {
        const int kk  = s >> 4;
        const int ic0 = (s & 15) << 5;
        const int dy  = kk / 3 - 1;
        const int dx  = kk % 3 - 1;
        #pragma unroll
        for (int i = 0; i < 8; ++i) {
            int p = i * 256 + tid;           // 0..2047 k-pairs
            int m = p & 127;
            int t = (p >> 7) << 1;           // even k
            int pos = m0 + m;
            int b  = pos >> 12;
            int y  = (pos >> 6) & 63;
            int x  = pos & 63;
            int yy = y + dy, xx = x + dx;
            bool ok = ((unsigned)yy < 64u) && ((unsigned)xx < 64u);
            int yyc = yy < 0 ? 0 : (yy > 63 ? 63 : yy);
            int xxc = xx < 0 ? 0 : (xx > 63 ? 63 : xx);
            size_t base = (((size_t)(b * 512 + ic0 + t) * 64 + yyc) * 64 + xxc);
            float f0 = feat[base];
            float f1 = feat[base + 4096];    // next ic plane
            v2bf pr;
            pr[0] = (__bf16)(ok ? f0 : 0.0f);
            pr[1] = (__bf16)(ok ? f1 : 0.0f);
            *(v2bf*)(&Ab[m * 40 + t]) = pr;
        }
    };
    // ---- B stage: contiguous 8KB tile -> LDS via async copy engine
    auto stageB = [&](int s, __bf16* Bb) {
        const __bf16* slab = Bpk + ((size_t)s * 512 + oc0) * 32;
        #pragma unroll
        for (int c2 = 0; c2 < 2; ++c2) {
            int chunk = c2 * 256 + tid;      // 512 x 16B chunks
            int n = chunk >> 2;
            int c = chunk & 3;
            async_g2l_b128(slab + n * 32 + c * 8, &Bb[n * 40 + c * 8]);
        }
    };

    // prologue: fill buffer 0
    stageA(0, As[0]);
    stageB(0, Bs[0]);
    wait_async0();
    __syncthreads();

    for (int s = 0; s < 144; ++s) {
        const int cur = s & 1, nxt = cur ^ 1;
        if (s + 1 < 144) {                   // overlap next-tile staging with math
            stageB(s + 1, Bs[nxt]);
            stageA(s + 1, As[nxt]);
        }
        v16bf bf[2];
        #pragma unroll
        for (int nt = 0; nt < 2; ++nt)
            bf[nt] = load_frag(&Bs[cur][(wn * 32 + nt * 16 + l15) * 40 + koff]);
        #pragma unroll
        for (int mt = 0; mt < 4; ++mt) {
            v16bf af = load_frag(&As[cur][(wm * 64 + mt * 16 + l15) * 40 + koff]);
            #pragma unroll
            for (int nt = 0; nt < 2; ++nt)
                acc[mt][nt] = __builtin_amdgcn_wmma_f32_16x16x32_bf16(
                    false, af, false, bf[nt], (short)0, acc[mt][nt], false, false);
        }
        wait_async0();
        __syncthreads();
    }

    // epilogue: bias + ReLU, store position-major bf16
    const int mbase = m0 + wm * 64 + ((lane & 16) ? 8 : 0);
    #pragma unroll
    for (int nt = 0; nt < 2; ++nt) {
        int oc = oc0 + wn * 32 + nt * 16 + l15;
        float bias = b_inter[oc];
        #pragma unroll
        for (int mt = 0; mt < 4; ++mt) {
            #pragma unroll
            for (int r = 0; r < 8; ++r) {
                int pos = mbase + mt * 16 + r;
                float v = fmaxf(acc[mt][nt][r] + bias, 0.0f);
                inter_t[(size_t)pos * 512 + oc] = (__bf16)v;
            }
        }
    }
}

// ---------------------------------------------------------------------------
// K2: fused 1x1 heads (cls sigmoid + reg) as WMMA GEMM, N padded to 48.
//   Weights async-copied to LDS; A fragments loaded straight from global.
// ---------------------------------------------------------------------------
__global__ __launch_bounds__(256) void rpn_head_wmma(
    const __bf16* __restrict__ inter_t, const __bf16* __restrict__ WcT,
    const float* __restrict__ b_cls, const float* __restrict__ b_reg,
    float* __restrict__ out_cls, float* __restrict__ reg_t) {
    __shared__ __attribute__((aligned(16))) __bf16 Ws[48 * 520];  // [n][512] +8 pad

    const int tid  = threadIdx.x;
    const int lane = tid & 31;
    const int wid  = tid >> 5;
    const int l15  = lane & 15;
    const int koff = (lane & 16) ? 8 : 0;

    #pragma unroll
    for (int c2 = 0; c2 < 12; ++c2) {       // 3072 x 16B chunks
        int chunk = c2 * 256 + tid;
        int n = chunk >> 6;
        int c = chunk & 63;
        async_g2l_b128(WcT + n * 512 + c * 8, &Ws[n * 520 + c * 8]);
    }
    wait_async0();
    __syncthreads();

    const int row = blockIdx.x * 128 + wid * 16 + l15;
    v8f acc[3];
    #pragma unroll
    for (int n = 0; n < 3; ++n)
        #pragma unroll
        for (int r = 0; r < 8; ++r) acc[n][r] = 0.0f;

    const __bf16* arow = inter_t + (size_t)row * 512 + koff;
    for (int k0 = 0; k0 < 512; k0 += 32) {
        v16bf af = load_frag(arow + k0);
        #pragma unroll
        for (int nt = 0; nt < 3; ++nt) {
            v16bf bf = load_frag(&Ws[(nt * 16 + l15) * 520 + k0 + koff]);
            acc[nt] = __builtin_amdgcn_wmma_f32_16x16x32_bf16(
                false, af, false, bf, (short)0, acc[nt], false, false);
        }
    }

    const int posb = blockIdx.x * 128 + wid * 16 + ((lane & 16) ? 8 : 0);
    #pragma unroll
    for (int nt = 0; nt < 3; ++nt) {
        int oc = nt * 16 + l15;
        #pragma unroll
        for (int r = 0; r < 8; ++r) {
            int pos = posb + r;
            float v = acc[nt][r];
            if (oc < 9) {
                v += b_cls[oc];
                float s = 1.0f / (1.0f + __expf(-v));
                int b = pos >> 12, hw = pos & 4095;
                out_cls[(b * 9 + oc) * 4096 + hw] = s;
            } else if (oc < 45) {
                int c = oc - 9;
                v += b_reg[c];
                reg_t[(size_t)pos * 36 + c] = v;
            }
        }
    }
}

// ---------------------------------------------------------------------------
// K3: box regression with inline anchors -> moved [B,KA,H,W,4]
// ---------------------------------------------------------------------------
__global__ __launch_bounds__(256) void rpn_boxreg(
    const float* __restrict__ reg_t, float* __restrict__ moved) {
    int i = blockIdx.x * 256 + threadIdx.x;
    if (i >= BB * KHW) return;
    int b  = i / KHW;
    int r  = i - b * KHW;
    int k  = r >> 12;
    int hw = r & 4095;
    int y  = hw >> 6, x = hw & 63;
    int pos = b * 4096 + hw;
    const float* t = reg_t + (size_t)pos * 36 + k * 4;
    float tx = t[0], ty = t[1], tw = t[2], th = t[3];
    int si = k / 3, ri = k - si * 3;
    float size  = 128.0f * (float)(1 << si);
    float ratio = (ri == 0) ? 0.5f : ((ri == 1) ? 1.0f : 2.0f);
    float sq = sqrtf(ratio);
    float aw = size / sq, ah = size * sq;
    float ax = (x + 0.5f) * 16.0f, ay = (y + 0.5f) * 16.0f;
    float* o = moved + (size_t)i * 4;
    o[0] = ax + aw * tx;
    o[1] = ay + ah * ty;
    o[2] = aw * __expf(tw);
    o[3] = ah * __expf(th);
}

// ---------------------------------------------------------------------------
// K4: per (b,gt): argmax_a IoU over anchors (block reduction, first-index ties)
// ---------------------------------------------------------------------------
__global__ __launch_bounds__(256) void rpn_best_anchor(
    const float* __restrict__ moved, const float* __restrict__ gt,
    int* __restrict__ best) {
    __shared__ float sio[256];
    __shared__ int   sid[256];
    int bg = blockIdx.x;
    int b = bg >> 6, g = bg & 63;
    const float* gp = gt + ((size_t)b * NGT + g) * 4;
    float gx = gp[0], gy = gp[1], gw = gp[2], gh = gp[3];
    float bi = -1e30f; int ba = 0;
    for (int a = threadIdx.x; a < KHW; a += 256) {
        const float* p = moved + ((size_t)b * KHW + a) * 4;
        float iou = iou_xywh(p[0], p[1], p[2], p[3], gx, gy, gw, gh);
        if (iou > bi) { bi = iou; ba = a; }
    }
    sio[threadIdx.x] = bi; sid[threadIdx.x] = ba;
    __syncthreads();
    for (int s = 128; s > 0; s >>= 1) {
        if (threadIdx.x < (unsigned)s) {
            float oi = sio[threadIdx.x + s]; int oa = sid[threadIdx.x + s];
            if (oi > sio[threadIdx.x] ||
                (oi == sio[threadIdx.x] && oa < sid[threadIdx.x])) {
                sio[threadIdx.x] = oi; sid[threadIdx.x] = oa;
            }
        }
        __syncthreads();
    }
    if (threadIdx.x == 0) best[bg] = sid[0];
}

// ---------------------------------------------------------------------------
// K5: per anchor: pos/neg thresholds + argmax over gts
// ---------------------------------------------------------------------------
__global__ __launch_bounds__(256) void rpn_labels(
    const float* __restrict__ moved, const float* __restrict__ gt,
    float* __restrict__ out_pos, float* __restrict__ out_neg,
    float* __restrict__ out_hgt) {
    int i = blockIdx.x * 256 + threadIdx.x;
    if (i >= BB * KHW) return;
    int b = i / KHW;
    const float* p = moved + (size_t)i * 4;
    float ax = p[0], ay = p[1], aw = p[2], ah = p[3];
    const float* gb = gt + (size_t)b * NGT * 4;
    bool pos_any = false, neg_any = false;
    float bi = -1e30f; int bg = 0;
    for (int g = 0; g < NGT; ++g) {
        const float* q = gb + g * 4;
        float iou = iou_xywh(ax, ay, aw, ah, q[0], q[1], q[2], q[3]);
        pos_any = pos_any || (iou > POS_T);
        neg_any = neg_any || (iou > NEG_T);
        if (iou > bi) { bi = iou; bg = g; }
    }
    out_pos[i] = pos_any ? 1.0f : 0.0f;
    out_neg[i] = neg_any ? 0.0f : 1.0f;
    out_hgt[i] = (float)bg;
}

// K6: force pos=1 for each (b,gt) winner anchor
__global__ __launch_bounds__(256) void rpn_mark_best(
    const int* __restrict__ best, float* __restrict__ out_pos) {
    int t = blockIdx.x * 256 + threadIdx.x;
    if (t >= BB * NGT) return;
    int b = t >> 6;
    out_pos[(size_t)b * KHW + best[t]] = 1.0f;
}

// ---------------------------------------------------------------------------
extern "C" void kernel_launch(void* const* d_in, const int* in_sizes, int n_in,
                              void* d_out, int out_size, void* d_ws, size_t ws_size,
                              hipStream_t stream) {
    const float* features = (const float*)d_in[0];
    const float* w_inter  = (const float*)d_in[1];
    const float* b_inter  = (const float*)d_in[2];
    const float* w_cls    = (const float*)d_in[3];
    const float* b_cls    = (const float*)d_in[4];
    const float* w_reg    = (const float*)d_in[5];
    const float* b_reg    = (const float*)d_in[6];
    const float* gt_bbox  = (const float*)d_in[7];

    float* out = (float*)d_out;
    char*  ws  = (char*)d_ws;
    __bf16* Bpk     = (__bf16*)(ws + WS_BW);
    __bf16* WcT     = (__bf16*)(ws + WS_WC);
    __bf16* inter_t = (__bf16*)(ws + WS_INTER);
    int*    best    = (int*)   (ws + WS_BEST);
    float*  reg_t   = (float*) (ws + WS_REGT);

    rpn_prep_weights<<<(4608 * 512 + 48 * 512) / 256, 256, 0, stream>>>(
        w_inter, w_cls, w_reg, Bpk, WcT);

    rpn_conv3_wmma<<<dim3(POSN / 128, 4), 256, 0, stream>>>(
        features, Bpk, b_inter, inter_t);

    rpn_head_wmma<<<POSN / 128, 256, 0, stream>>>(
        inter_t, WcT, b_cls, b_reg, out + CLS_OFF, reg_t);

    rpn_boxreg<<<(BB * KHW) / 256, 256, 0, stream>>>(reg_t, out + MOVED_OFF);

    rpn_best_anchor<<<BB * NGT, 256, 0, stream>>>(out + MOVED_OFF, gt_bbox, best);

    rpn_labels<<<(BB * KHW) / 256, 256, 0, stream>>>(
        out + MOVED_OFF, gt_bbox, out + POSL_OFF, out + NEGL_OFF, out + HGT_OFF);

    rpn_mark_best<<<2, 256, 0, stream>>>(best, out + POSL_OFF);
}
